// SemVecToMelModel_35957466202505
// MI455X (gfx1250) — matrix operand order repaired and checked by reading, not compiled, not run on hardware
//
#include <hip/hip_runtime.h>

// ---------------- model dims ----------------
#define B_    32
#define S_    512
#define IN_   300
#define H_    180
#define H4_   720
#define OUT_  60
#define L_    4

#define KPAD0 320      // IN padded to mult of 32
#define KS0   10       // 320/32 k-steps
#define KPADH 192      // H padded to mult of 32
#define KSH   6        // 192/32 k-steps
#define NT4H  45       // 720/16 n-tiles
#define GROW  728      // LDS gate row stride (floats), bank-skewed

typedef __attribute__((ext_vector_type(16))) __bf16 v16bf;
typedef __attribute__((ext_vector_type(8)))  float  v8f;

union FragB {
  v16bf v;
  uint4 q[2];
};

__device__ __forceinline__ unsigned short f2bf(float f) {
  unsigned int u = __builtin_bit_cast(unsigned int, f);
  u += 0x7fffu + ((u >> 16) & 1u);          // round-to-nearest-even
  return (unsigned short)(u >> 16);
}

// A fragment: 16x32 bf16, row-major source with row stride lda (elements).
// Lanes 0-15: M=lane, elems0-7 = K 0..7, elems8-15 = K 16..23
// Lanes16-31: M=lane-16, elems0-7 = K 8..15, elems8-15 = K 24..31
__device__ __forceinline__ void load_a(FragB& f, const unsigned short* A, int lda, int lane) {
  int m  = lane & 15;
  int kb = (lane < 16) ? 0 : 8;
  f.q[0] = *(const uint4*)(A + m * lda + kb);
  f.q[1] = *(const uint4*)(A + m * lda + kb + 16);
}

// B fragment from fragment-major packed buffer: 32 lanes x 16 bf16 contiguous.
__device__ __forceinline__ void load_b(FragB& f, const unsigned short* P, int lane) {
  const uint4* p = (const uint4*)(P + lane * 16);
  f.q[0] = p[0];
  f.q[1] = p[1];
}

__device__ __forceinline__ v8f wmma_bf16(const FragB& a, const FragB& b, v8f c) {
  return __builtin_amdgcn_wmma_f32_16x16x32_bf16(false, a.v, false, b.v,
                                                 (short)0, c, false, false);
}

// ---------------- packing kernels ----------------
// A0[m][k] = bf16(x[b][t][k]), m = t*32+b, k padded to 320
__global__ void pack_a0(const float* __restrict__ x, unsigned short* __restrict__ A0) {
  int idx = blockIdx.x * blockDim.x + threadIdx.x;   // < 16384*320
  int m = idx / KPAD0, k = idx % KPAD0;
  int t = m >> 5, b = m & 31;
  float v = (k < IN_) ? x[((size_t)b * S_ + t) * IN_ + k] : 0.f;
  A0[idx] = f2bf(v);
}

// Pack W[N][K] row-major fp32 into WMMA-B fragment-major bf16:
// P[((nt*KS+ks)*32 + lane)*16 + e] = W[nt*16+(lane&15)][ks*32+(lane<16?0:8)+(e&7)+(e&8?16:0)]
__global__ void pack_bmat(const float* __restrict__ W, unsigned short* __restrict__ P,
                          int N, int K, int KS) {
  int idx  = blockIdx.x * blockDim.x + threadIdx.x;  // NT*KS*512
  int e    = idx & 15;
  int lane = (idx >> 4) & 31;
  int rest = idx >> 9;
  int ks   = rest % KS;
  int nt   = rest / KS;
  int n = nt * 16 + (lane & 15);
  int k = ks * 32 + ((lane < 16) ? 0 : 8) + (e & 7) + ((e & 8) ? 16 : 0);
  float v = (n < N && k < K) ? W[(size_t)n * K + k] : 0.f;
  P[idx] = f2bf(v);
}

__global__ void bias_combine(const float* __restrict__ bih, const float* __restrict__ bhh,
                             float* __restrict__ biasC) {
  int i = blockIdx.x * blockDim.x + threadIdx.x;
  if (i < L_ * H4_) biasC[i] = bih[i] + bhh[i];
}

// ---------------- layer-0 input projection: XG0 = A0 @ Wih0^T + b0 ----------------
// grid 5760 x 256 = 46080 waves = 1024 m-tiles * 45 n-tiles (exact, EXEC all-ones)
__global__ __launch_bounds__(256) void gemm_xg0(const unsigned short* __restrict__ A0,
                                                const unsigned short* __restrict__ W0P,
                                                const float* __restrict__ biasC,
                                                float* __restrict__ XG0) {
  int lane = threadIdx.x & 31;
  int wave = blockIdx.x * (blockDim.x >> 5) + (threadIdx.x >> 5);
  int mt = wave / NT4H, nt = wave % NT4H;
  int col = nt * 16 + (lane & 15);
  float bv = biasC[col];
  v8f acc;
  for (int r = 0; r < 8; ++r) acc[r] = bv;
  const unsigned short* Abase = A0 + (size_t)(mt * 16) * KPAD0;
  const unsigned short* Pbase = W0P + (size_t)(nt * KS0) * 512;
  for (int ks = 0; ks < KS0; ++ks) {
    FragB a, b;
    load_a(a, Abase + ks * 32, KPAD0, lane);
    load_b(b, Pbase + (size_t)ks * 512, lane);
    acc = wmma_bf16(a, b, acc);
  }
  int rb = mt * 16 + ((lane >= 16) ? 8 : 0);
  for (int r = 0; r < 8; ++r)
    XG0[(size_t)(rb + r) * H4_ + col] = acc[r];
}

// ---------------- pipelined LSTM scan: 4 persistent blocks, 1 per layer ----------------
// LDS: h bf16[32][192] + c f32[32][192] + gates f32[32][GROW]  (~127 KB / 320 KB)
__global__ __launch_bounds__(1024) void lstm_scan(const float* __restrict__ XG0,
                                                  const unsigned short* __restrict__ WihP,
                                                  const unsigned short* __restrict__ WhhP,
                                                  const float* __restrict__ biasC,
                                                  unsigned short* __restrict__ Hpub,
                                                  int* progress) {
  const int l = blockIdx.x;
  extern __shared__ char smem[];
  unsigned short* h_s = (unsigned short*)smem;                       // 32*192 bf16
  float* c_s = (float*)(smem + B_ * KPADH * 2);                      // 32*192 f32
  float* g_s = (float*)(smem + B_ * KPADH * 2 + B_ * KPADH * 4);     // 32*GROW f32
  const int tid  = threadIdx.x;
  const int lane = tid & 31;
  const int wave = tid >> 5;
  for (int i = tid; i < B_ * KPADH; i += 1024) { h_s[i] = 0; c_s[i] = 0.f; }
  const unsigned short* whh = WhhP + (size_t)l * (NT4H * KSH * 512);
  const unsigned short* wih = WihP + (size_t)(l > 0 ? l - 1 : 0) * (NT4H * KSH * 512);
  const float* bias = biasC + l * H4_;
  for (int t = 0; t < S_; ++t) {
    if (l > 0 && tid == 0) {
      while (__hip_atomic_load(&progress[l - 1], __ATOMIC_ACQUIRE,
                               __HIP_MEMORY_SCOPE_AGENT) <= t)
        __builtin_amdgcn_s_sleep(1);
    }
    __syncthreads();
    // gate matrix g[32][720] = xg(t) + h(t-1) @ Whh^T : 90 tiles over 32 waves
    for (int tile = wave; tile < 2 * NT4H; tile += 32) {
      int mt = tile / NT4H, nt = tile % NT4H;
      int m0 = mt * 16, n0 = nt * 16;
      int col  = n0 + (lane & 15);
      int rsel = (lane >= 16) ? 8 : 0;
      v8f acc;
      if (l == 0) {                                    // xg precomputed (bias included)
        const float* xg = XG0 + (size_t)(t * B_ + m0 + rsel) * H4_ + col;
        for (int r = 0; r < 8; ++r) acc[r] = xg[(size_t)r * H4_];
      } else {                                         // incremental input projection
        float bv = bias[col];
        for (int r = 0; r < 8; ++r) acc[r] = bv;
        const unsigned short* hprev = Hpub + ((size_t)(l - 1) * S_ + t) * (B_ * KPADH);
        const unsigned short* P = wih + (size_t)(nt * KSH) * 512;
        __builtin_prefetch(P, 0, 1);
        for (int ks = 0; ks < KSH; ++ks) {
          FragB a, b;
          load_a(a, hprev + m0 * KPADH + ks * 32, KPADH, lane);
          load_b(b, P + (size_t)ks * 512, lane);
          acc = wmma_bf16(a, b, acc);
        }
      }
      const unsigned short* P = whh + (size_t)(nt * KSH) * 512;
      __builtin_prefetch(P, 0, 1);
      for (int ks = 0; ks < KSH; ++ks) {
        FragB a, b;
        load_a(a, h_s + m0 * KPADH + ks * 32, KPADH, lane);   // LDS-resident h
        load_b(b, P + (size_t)ks * 512, lane);
        acc = wmma_bf16(a, b, acc);
      }
      for (int r = 0; r < 8; ++r)
        g_s[(m0 + rsel + r) * GROW + col] = acc[r];
    }
    __syncthreads();
    // elementwise LSTM cell (gate order i,f,g,o), publish h(t) as bf16
    unsigned short* hout = Hpub + ((size_t)l * S_ + t) * (B_ * KPADH);
    for (int idx = tid; idx < B_ * KPADH; idx += 1024) {
      int b = idx / KPADH, j = idx % KPADH;
      unsigned short hb = 0;
      if (j < H_) {
        float gi = g_s[b * GROW + j];
        float gf = g_s[b * GROW + H_ + j];
        float gg = g_s[b * GROW + 2 * H_ + j];
        float go = g_s[b * GROW + 3 * H_ + j];
        float si = 1.f / (1.f + __expf(-gi));
        float sf = 1.f / (1.f + __expf(-gf));
        float tg = tanhf(gg);
        float so = 1.f / (1.f + __expf(-go));
        float c  = sf * c_s[idx] + si * tg;
        c_s[idx] = c;
        hb = f2bf(so * tanhf(c));
      }
      h_s[idx]  = hb;
      hout[idx] = hb;
    }
    __threadfence();
    __syncthreads();
    if (tid == 0)
      __hip_atomic_store(&progress[l], t + 1, __ATOMIC_RELEASE,
                         __HIP_MEMORY_SCOPE_AGENT);
  }
}

// ---------------- final linear: [16384,192] @ [192,64] -> lstm_out[B][60][S] ----------------
// grid 512 x 256 = 4096 waves = 1024 m-tiles * 4 n-tiles (exact)
__global__ __launch_bounds__(256) void linear_out(const unsigned short* __restrict__ H3,
                                                  const unsigned short* __restrict__ LinP,
                                                  const float* __restrict__ linb,
                                                  float* __restrict__ lstm_o) {
  int lane = threadIdx.x & 31;
  int wave = blockIdx.x * (blockDim.x >> 5) + (threadIdx.x >> 5);
  int mt = wave >> 2, nt = wave & 3;
  int col = nt * 16 + (lane & 15);
  float bv = (col < OUT_) ? linb[col] : 0.f;
  v8f acc;
  for (int r = 0; r < 8; ++r) acc[r] = bv;
  const unsigned short* Abase = H3 + (size_t)(mt * 16) * KPADH;
  const unsigned short* Pbase = LinP + (size_t)(nt * KSH) * 512;
  for (int ks = 0; ks < KSH; ++ks) {
    FragB a, b;
    load_a(a, Abase + ks * 32, KPADH, lane);
    load_b(b, Pbase + (size_t)ks * 512, lane);
    acc = wmma_bf16(a, b, acc);
  }
  if (col < OUT_) {
    int rb = mt * 16 + ((lane >= 16) ? 8 : 0);
    for (int r = 0; r < 8; ++r) {
      int m = rb + r;
      int t = m >> 5, b = m & 31;
      lstm_o[((size_t)b * OUT_ + col) * S_ + t] = acc[r];
    }
  }
}

// ---------------- mel-smoothing block (shifted grouped conv + residual) ----------------
__global__ void mel_block(const float* __restrict__ xin, const float* __restrict__ w,
                          const float* __restrict__ bias, float* __restrict__ xout) {
  int bc = blockIdx.x;
  int bb = bc / OUT_, c = bc % OUT_;
  int i = c / 3, j = c % 3;       // output channel c = 3*i + j
  float wr[15];
  for (int q = 0; q < 15; ++q) wr[q] = w[(j * 20 + i) * 15 + q];
  float bv = bias[j * 20 + i];
  for (int t = threadIdx.x; t < S_; t += blockDim.x) {
    float acc = bv + xin[((size_t)bb * OUT_ + c) * S_ + t];   // residual shortcut
    for (int cc = 0; cc < 3; ++cc) {
      int ch = 3 * i + cc + j - 1;          // channel-shifted stack xs[j]
      if (ch < 0 || ch >= OUT_) continue;
      const float* row = xin + ((size_t)bb * OUT_ + ch) * S_;
      for (int kk = 0; kk < 5; ++kk) {
        int tt = t + kk - 2;
        if (tt >= 0 && tt < S_) acc += wr[cc * 5 + kk] * row[tt];
      }
    }
    xout[((size_t)bb * OUT_ + c) * S_ + t] = acc;
  }
}

// ---------------- residual grouped conv over interleaved (lstm, smoothed) ----------------
__global__ void res_out(const float* __restrict__ a, const float* __restrict__ s,
                        const float* __restrict__ rw, const float* __restrict__ rb,
                        float* __restrict__ out) {
  int bc = blockIdx.x;
  int bb = bc / OUT_, c = bc % OUT_;
  float w0[5], w1[5];
  for (int kk = 0; kk < 5; ++kk) {
    w0[kk] = rw[(c * 2 + 0) * 5 + kk];
    w1[kk] = rw[(c * 2 + 1) * 5 + kk];
  }
  const float* r0 = a + ((size_t)bb * OUT_ + c) * S_;
  const float* r1 = s + ((size_t)bb * OUT_ + c) * S_;
  float bv = rb[c];
  for (int t = threadIdx.x; t < S_; t += blockDim.x) {
    float acc = bv;
    for (int kk = 0; kk < 5; ++kk) {
      int tt = t + kk - 2;
      if (tt >= 0 && tt < S_) acc += w0[kk] * r0[tt] + w1[kk] * r1[tt];
    }
    out[((size_t)bb * S_ + t) * OUT_ + c] = acc;   // [B,S,60]
  }
}

extern "C" void kernel_launch(void* const* d_in, const int* in_sizes, int n_in,
                              void* d_out, int out_size, void* d_ws, size_t ws_size,
                              hipStream_t stream) {
  (void)in_sizes; (void)n_in; (void)out_size; (void)ws_size;
  const float* x    = (const float*)d_in[0];
  const float* Wih0 = (const float*)d_in[1];
  const float* WihR = (const float*)d_in[2];
  const float* Whh  = (const float*)d_in[3];
  const float* bih  = (const float*)d_in[4];
  const float* bhh  = (const float*)d_in[5];
  const float* linW = (const float*)d_in[6];
  const float* linb = (const float*)d_in[7];
  const float* melw = (const float*)d_in[8];
  const float* melb = (const float*)d_in[9];
  const float* resW = (const float*)d_in[10];
  const float* resb = (const float*)d_in[11];
  float* outp = (float*)d_out;

  char* ws = (char*)d_ws;
  size_t off = 0;
  auto alloc = [&](size_t bytes) -> char* {
    char* p = ws + off;
    off += (bytes + 255) & ~(size_t)255;
    return p;
  };
  unsigned short* A0   = (unsigned short*)alloc((size_t)16384 * KPAD0 * 2);
  unsigned short* W0P  = (unsigned short*)alloc((size_t)NT4H * KS0 * 512 * 2);
  unsigned short* WihP = (unsigned short*)alloc((size_t)3 * NT4H * KSH * 512 * 2);
  unsigned short* WhhP = (unsigned short*)alloc((size_t)4 * NT4H * KSH * 512 * 2);
  unsigned short* LinP = (unsigned short*)alloc((size_t)4 * KSH * 512 * 2);
  float* biasC = (float*)alloc((size_t)L_ * H4_ * 4);
  float* XG0   = (float*)alloc((size_t)16384 * H4_ * 4);
  unsigned short* Hpub = (unsigned short*)alloc((size_t)L_ * S_ * B_ * KPADH * 2);
  float* lstm_o = (float*)alloc((size_t)B_ * OUT_ * S_ * 4);
  float* mAbuf  = (float*)alloc((size_t)B_ * OUT_ * S_ * 4);
  float* mBbuf  = (float*)alloc((size_t)B_ * OUT_ * S_ * 4);
  int* progress = (int*)alloc(256);

  hipMemsetAsync(progress, 0, 256, stream);
  pack_a0<<<(16384 * KPAD0) / 256, 256, 0, stream>>>(x, A0);
  pack_bmat<<<(NT4H * KS0 * 512) / 256, 256, 0, stream>>>(Wih0, W0P, H4_, IN_, KS0);
  for (int l = 0; l < 3; ++l)
    pack_bmat<<<(NT4H * KSH * 512) / 256, 256, 0, stream>>>(
        WihR + (size_t)l * H4_ * H_, WihP + (size_t)l * NT4H * KSH * 512, H4_, H_, KSH);
  for (int l = 0; l < 4; ++l)
    pack_bmat<<<(NT4H * KSH * 512) / 256, 256, 0, stream>>>(
        Whh + (size_t)l * H4_ * H_, WhhP + (size_t)l * NT4H * KSH * 512, H4_, H_, KSH);
  pack_bmat<<<(4 * KSH * 512) / 256, 256, 0, stream>>>(linW, LinP, OUT_, H_, KSH);
  bias_combine<<<(L_ * H4_ + 255) / 256, 256, 0, stream>>>(bih, bhh, biasC);

  gemm_xg0<<<(1024 * NT4H) / 8, 256, 0, stream>>>(A0, W0P, biasC, XG0);

  size_t smem = (size_t)B_ * KPADH * 2 + (size_t)B_ * KPADH * 4 + (size_t)B_ * GROW * 4;
  (void)hipFuncSetAttribute((const void*)lstm_scan,
                            hipFuncAttributeMaxDynamicSharedMemorySize, (int)smem);
  lstm_scan<<<4, 1024, smem, stream>>>(XG0, WihP, WhhP, biasC, Hpub, progress);

  linear_out<<<(1024 * 4) / 8, 256, 0, stream>>>(
      Hpub + (size_t)3 * S_ * B_ * KPADH, LinP, linb, lstm_o);

  mel_block<<<B_ * OUT_, 256, 0, stream>>>(lstm_o, melw,        melb,       mAbuf);
  mel_block<<<B_ * OUT_, 256, 0, stream>>>(mAbuf,  melw + 900,  melb + 60,  mBbuf);
  mel_block<<<B_ * OUT_, 256, 0, stream>>>(mBbuf,  melw + 1800, melb + 120, mAbuf);
  res_out<<<B_ * OUT_, 256, 0, stream>>>(lstm_o, mAbuf, resW, resb, outp);
}
